// SparseMoEBlock_9328668967127
// MI455X (gfx1250) — compile-verified, matrix operand order checked by reading, not run on hardware
//
#include <hip/hip_runtime.h>

#define Bn 4
#define Ln 1024
#define Dn 1024
#define En 8
#define Hn 4096
#define CAP 6
#define NT (Bn * Ln)          // 4096 tokens
#define KSEL (NT * CAP)       // 24576 -> kth smallest rank
#define TILES_PER_E (NT / 16) // 256

typedef __attribute__((ext_vector_type(16))) __bf16 v16bf;
typedef __attribute__((ext_vector_type(8)))  float  v8f;

static __device__ inline v8f wmma_bf16(v16bf a, v16bf b, v8f c) {
  return __builtin_amdgcn_wmma_f32_16x16x32_bf16(false, a, false, b, (short)0, c,
                                                 false, false);
}

static __device__ inline float gelu_tanh(float v) {
  return 0.5f * v * (1.0f + tanhf(0.7978845608028654f * (v + 0.044715f * v * v * v)));
}

// A/B 16-bit fragment per-lane layout (ISA 7.12.2): lane holds K runs
// [kk + hi*8, +8) -> elements 0..7 and [kk + 16 + hi*8, +8) -> elements 8..15.
static __device__ inline v16bf load_frag_lds(const __bf16* row, int kk, int hi) {
  v16bf v;
  const __bf16* q0 = row + kk + hi * 8;
  const __bf16* q1 = row + kk + 16 + hi * 8;
#pragma unroll
  for (int j = 0; j < 8; ++j) { v[j] = q0[j]; v[8 + j] = q1[j]; }
  return v;
}

static __device__ inline v16bf load_frag_gf32(const float* p, int hi) {
  v16bf v;
  const float4 a0 = *reinterpret_cast<const float4*>(p + hi * 8);
  const float4 a1 = *reinterpret_cast<const float4*>(p + hi * 8 + 4);
  const float4 c0 = *reinterpret_cast<const float4*>(p + 16 + hi * 8);
  const float4 c1 = *reinterpret_cast<const float4*>(p + 16 + hi * 8 + 4);
  v[0] = (__bf16)a0.x; v[1] = (__bf16)a0.y; v[2] = (__bf16)a0.z; v[3] = (__bf16)a0.w;
  v[4] = (__bf16)a1.x; v[5] = (__bf16)a1.y; v[6] = (__bf16)a1.z; v[7] = (__bf16)a1.w;
  v[8] = (__bf16)c0.x; v[9] = (__bf16)c0.y; v[10] = (__bf16)c0.z; v[11] = (__bf16)c0.w;
  v[12] = (__bf16)c1.x; v[13] = (__bf16)c1.y; v[14] = (__bf16)c1.z; v[15] = (__bf16)c1.w;
  return v;
}

static __device__ inline v16bf load_frag_gbf16(const __bf16* p, int hi) {
  union { uint4 u; __bf16 h[8]; } c0, c1;
  c0.u = *reinterpret_cast<const uint4*>(p + hi * 8);
  c1.u = *reinterpret_cast<const uint4*>(p + 16 + hi * 8);
  v16bf v;
#pragma unroll
  for (int j = 0; j < 8; ++j) { v[j] = c0.h[j]; v[8 + j] = c1.h[j]; }
  return v;
}

// ---------------- K0: zero output ----------------
__global__ void moe_zero_kernel(float* p, int n) {
  for (int i = blockIdx.x * blockDim.x + threadIdx.x; i < n; i += gridDim.x * blockDim.x)
    p[i] = 0.0f;
}

// ---------------- K0b: fp32 -> bf16 bulk convert ----------------
__global__ __launch_bounds__(256) void moe_cvt_kernel(const float* __restrict__ in,
                                                      __bf16* __restrict__ out, long n) {
  long i = ((long)blockIdx.x * blockDim.x + threadIdx.x) * 8;
  long stride = (long)gridDim.x * blockDim.x * 8;
  for (; i + 7 < n; i += stride) {
    float4 a = *reinterpret_cast<const float4*>(in + i);
    float4 b = *reinterpret_cast<const float4*>(in + i + 4);
    union { uint4 u; __bf16 h[8]; } o;
    o.h[0] = (__bf16)a.x; o.h[1] = (__bf16)a.y; o.h[2] = (__bf16)a.z; o.h[3] = (__bf16)a.w;
    o.h[4] = (__bf16)b.x; o.h[5] = (__bf16)b.y; o.h[6] = (__bf16)b.z; o.h[7] = (__bf16)b.w;
    *reinterpret_cast<uint4*>(out + i) = o.u;
  }
}

// ---------------- K1: router logits [NT, E] ----------------
__global__ __launch_bounds__(256) void moe_router_kernel(const float* __restrict__ x,
                                                         const float* __restrict__ gw,
                                                         float* __restrict__ logits) {
  int idx = blockIdx.x * blockDim.x + threadIdx.x;
  if (idx >= NT * En) return;
  int t = idx >> 3, e = idx & 7;
  const float* xr = x + (size_t)t * Dn;
  const float* gr = gw + (size_t)e * Dn;
  float acc = 0.0f;
  for (int i = 0; i < Dn; i += 4) {
    float4 a = *reinterpret_cast<const float4*>(xr + i);
    float4 b = *reinterpret_cast<const float4*>(gr + i);
    acc += a.x * b.x + a.y * b.y + a.z * b.z + a.w * b.w;
  }
  logits[idx] = acc;
}

// ---------------- K2: exact kth-smallest via 4-pass radix select ----------------
__global__ __launch_bounds__(256) void moe_kth_kernel(const float* __restrict__ logits,
                                                      float* __restrict__ kth_out) {
  const int n = NT * En;
  __shared__ unsigned hist[256];
  __shared__ unsigned sprefix;
  __shared__ int srank;
  int tid = threadIdx.x;
  if (tid == 0) { sprefix = 0u; srank = KSEL - 1; }
  __syncthreads();
  for (int pass = 3; pass >= 0; --pass) {
    int shift = pass * 8;
    hist[tid] = 0u;
    __syncthreads();
    unsigned prefix = sprefix;
    unsigned maskhi = (pass == 3) ? 0u : (0xFFFFFFFFu << (8 * (pass + 1)));
    for (int i = tid; i < n; i += 256) {
      unsigned bits = __float_as_uint(logits[i]);
      unsigned u = (bits & 0x80000000u) ? ~bits : (bits | 0x80000000u);
      if ((u & maskhi) == (prefix & maskhi)) atomicAdd(&hist[(u >> shift) & 0xFFu], 1u);
    }
    __syncthreads();
    if (tid == 0) {
      unsigned cum = 0; int r = srank; unsigned bfound = 255u;
      for (unsigned b = 0; b < 256u; ++b) {
        unsigned h = hist[b];
        if ((unsigned)r < cum + h) { bfound = b; r -= (int)cum; break; }
        cum += h;
      }
      sprefix = prefix | (bfound << shift);
      srank = r;
    }
    __syncthreads();
  }
  if (tid == 0) {
    unsigned u = sprefix;
    unsigned bits = (u & 0x80000000u) ? (u ^ 0x80000000u) : ~u;
    *kth_out = __uint_as_float(bits);
  }
}

// ---------------- K3: per-expert deterministic token compaction ----------------
__global__ __launch_bounds__(256) void moe_lists_kernel(const float* __restrict__ logits,
                                                        const float* __restrict__ kth,
                                                        int* __restrict__ lists,
                                                        int* __restrict__ counts) {
  int e = blockIdx.x, tid = threadIdx.x;
  __shared__ int cnts[257];
  float kv = *kth;
  const int TPT = NT / 256; // 16 tokens per thread, in token order
  int base = tid * TPT;
  int local = 0;
  for (int i = 0; i < TPT; ++i)
    if (logits[(base + i) * En + e] >= kv) ++local;
  cnts[tid] = local;
  __syncthreads();
  if (tid == 0) {
    int acc = 0;
    for (int i = 0; i < 256; ++i) { int c = cnts[i]; cnts[i] = acc; acc += c; }
    counts[e] = acc;
  }
  __syncthreads();
  int w = cnts[tid];
  for (int i = 0; i < TPT; ++i) {
    int t = base + i;
    if (logits[t * En + e] >= kv) lists[e * NT + (w++)] = t;
  }
}

// ---------------- K4: expert-grouped WMMA MLP ----------------
// PRE=1: weights/x pre-converted to bf16 in workspace (L2-resident, no hot-loop cvt)
// PRE=0: convert fp32 weights to bf16 on the fly
template <int PRE>
__global__ __launch_bounds__(256) void moe_gemm_kernel(
    const float* __restrict__ x, const __bf16* __restrict__ xbf,
    const float* __restrict__ w1, const __bf16* __restrict__ w1bf,
    const float* __restrict__ b1,
    const float* __restrict__ w2, const __bf16* __restrict__ w2bf,
    const float* __restrict__ b2,
    const float* __restrict__ logits, const int* __restrict__ lists,
    const int* __restrict__ counts, float* __restrict__ out) {
  int e = blockIdx.x >> 8;          // 256 tiles per expert
  int tile = blockIdx.x & 255;
  int ne = counts[e];
  if (tile * 16 >= ne) return;      // block-uniform exit: EXEC stays all-ones

  __shared__ __bf16 xs[16][1032];   // 16 tokens x 1024 (bf16), padded
  __shared__ __bf16 hs[16][136];    // 16 tokens x 128 hidden chunk (bf16), padded
  __shared__ float gateS[16];
  __shared__ int tokS[16];

  int tid = threadIdx.x;
  int wave = tid >> 5, lane = tid & 31, hi = lane >> 4, ln = lane & 15;

  if (tid < 16) {
    int idx = tile * 16 + tid;
    int t = (idx < ne) ? lists[e * NT + idx] : lists[e * NT];
    tokS[tid] = t;
    gateS[tid] = (idx < ne) ? logits[t * En + e] : 0.0f;
  }
  __syncthreads();

  { // stage x tile -> LDS bf16 (16 threads per row, 64 elems each)
    int row = tid >> 4;
    int c0 = (tid & 15) * 64;
    if (PRE) {
      const __bf16* xr = xbf + (size_t)tokS[row] * Dn + c0;
#pragma unroll
      for (int j = 0; j < 64; j += 8)
        *reinterpret_cast<uint4*>(&xs[row][c0 + j]) =
            *reinterpret_cast<const uint4*>(xr + j);
    } else {
      const float* xr = x + (size_t)tokS[row] * Dn + c0;
#pragma unroll 4
      for (int j = 0; j < 64; j += 4) {
        float4 f = *reinterpret_cast<const float4*>(xr + j);
        xs[row][c0 + j + 0] = (__bf16)f.x;
        xs[row][c0 + j + 1] = (__bf16)f.y;
        xs[row][c0 + j + 2] = (__bf16)f.z;
        xs[row][c0 + j + 3] = (__bf16)f.w;
      }
    }
  }
  __syncthreads();

  v8f accD[8];
#pragma unroll
  for (int j = 0; j < 8; ++j) accD[j] = (v8f){};

  for (int hc = 0; hc < Hn; hc += 128) {
    // GEMM1: hidden tile (16 tokens x 16 h-cols per wave) over K = D = 1024
    int h0 = hc + wave * 16;
    v8f hid = (v8f){};
    const float*  w1cf = w1   + ((size_t)e * Hn + (size_t)(h0 + ln)) * Dn;
    const __bf16* w1cb = w1bf + ((size_t)e * Hn + (size_t)(h0 + ln)) * Dn;
#pragma unroll 4
    for (int kk = 0; kk < Dn; kk += 32) {
      v16bf a = load_frag_lds(&xs[ln][0], kk, hi);
      v16bf b = PRE ? load_frag_gbf16(w1cb + kk, hi) : load_frag_gf32(w1cf + kk, hi);
      hid = wmma_bf16(a, b, hid);
    }
    float bias = b1[e * Hn + h0 + ln];
#pragma unroll
    for (int r = 0; r < 8; ++r) {
      int m = r + hi * 8;
      hs[m][wave * 16 + ln] = (__bf16)gelu_tanh(hid[r] + bias);
    }
    __syncthreads();

    // GEMM2: out[16 x 128 cols per wave] += hidden(16x128) * w2^T chunk.
    // Hoist the 4 A-fragments (K=128) and reuse across all 8 column tiles.
    v16bf afr[4];
#pragma unroll
    for (int kq = 0; kq < 4; ++kq) afr[kq] = load_frag_lds(&hs[ln][0], kq * 32, hi);
#pragma unroll
    for (int j = 0; j < 8; ++j) {
      int d0 = wave * 128 + j * 16;
      const float*  w2cf = w2   + ((size_t)e * Dn + (size_t)(d0 + ln)) * Hn + hc;
      const __bf16* w2cb = w2bf + ((size_t)e * Dn + (size_t)(d0 + ln)) * Hn + hc;
      v8f c = accD[j];
#pragma unroll
      for (int kq = 0; kq < 4; ++kq) {
        v16bf b = PRE ? load_frag_gbf16(w2cb + kq * 32, hi)
                      : load_frag_gf32(w2cf + kq * 32, hi);
        c = wmma_bf16(afr[kq], b, c);
      }
      accD[j] = c;
    }
    __syncthreads(); // hs reused next chunk
  }

  // epilogue: out[t, d] += gate * (acc + b2[e, d])
#pragma unroll
  for (int j = 0; j < 8; ++j) {
    int d0 = wave * 128 + j * 16;
    float b2v = b2[e * Dn + d0 + ln];
#pragma unroll
    for (int r = 0; r < 8; ++r) {
      int m = r + hi * 8;
      if (tile * 16 + m < ne) {
        float val = (accD[j][r] + b2v) * gateS[m];
        atomicAdd(&out[(size_t)tokS[m] * Dn + d0 + ln], val);
      }
    }
  }
}

extern "C" void kernel_launch(void* const* d_in, const int* in_sizes, int n_in,
                              void* d_out, int out_size, void* d_ws, size_t ws_size,
                              hipStream_t stream) {
  const float* x  = (const float*)d_in[0];
  const float* gw = (const float*)d_in[1];
  const float* w1 = (const float*)d_in[2];
  const float* b1 = (const float*)d_in[3];
  const float* w2 = (const float*)d_in[4];
  const float* b2 = (const float*)d_in[5];
  float* out = (float*)d_out;

  // workspace layout
  char* ws = (char*)d_ws;
  float* logits = (float*)ws;                         // NT*E floats (128 KB)
  float* kth    = logits + NT * En;
  int*   counts = (int*)(kth + 16);
  int*   lists  = counts + 16;                        // E*NT ints (128 KB)

  const size_t OFF_XBF  = (size_t)1 << 20;            // 1 MB
  const size_t OFF_W1BF = OFF_XBF  + (size_t)NT * Dn * 2;
  const size_t OFF_W2BF = OFF_W1BF + (size_t)En * Hn * Dn * 2;
  const size_t NEED     = OFF_W2BF + (size_t)En * Dn * Hn * 2;
  __bf16* xbf  = (__bf16*)(ws + OFF_XBF);
  __bf16* w1bf = (__bf16*)(ws + OFF_W1BF);
  __bf16* w2bf = (__bf16*)(ws + OFF_W2BF);
  const bool pre = ws_size >= NEED;                   // constant across calls

  moe_zero_kernel<<<2048, 256, 0, stream>>>(out, out_size);
  moe_router_kernel<<<(NT * En + 255) / 256, 256, 0, stream>>>(x, gw, logits);
  moe_kth_kernel<<<1, 256, 0, stream>>>(logits, kth);
  moe_lists_kernel<<<En, 256, 0, stream>>>(logits, kth, lists, counts);

  if (pre) {
    moe_cvt_kernel<<<1024, 256, 0, stream>>>(x,  xbf,  (long)NT * Dn);
    moe_cvt_kernel<<<4096, 256, 0, stream>>>(w1, w1bf, (long)En * Hn * Dn);
    moe_cvt_kernel<<<4096, 256, 0, stream>>>(w2, w2bf, (long)En * Dn * Hn);
    moe_gemm_kernel<1><<<En * TILES_PER_E, 256, 0, stream>>>(
        x, xbf, w1, w1bf, b1, w2, w2bf, b2, logits, lists, counts, out);
  } else {
    moe_gemm_kernel<0><<<En * TILES_PER_E, 256, 0, stream>>>(
        x, xbf, w1, w1bf, b1, w2, w2bf, b2, logits, lists, counts, out);
  }
}
